// EnsembleDecoder_61572651155468
// MI455X (gfx1250) — matrix operand order repaired and checked by reading, not compiled, not run on hardware
//
#include <hip/hip_runtime.h>
#include <hip/hip_bf16.h>
#include <math.h>

typedef __attribute__((ext_vector_type(16))) _Float16 v16h;
typedef __attribute__((ext_vector_type(8)))  _Float16 v8h;
typedef __attribute__((ext_vector_type(8)))  float    v8f;
typedef __attribute__((ext_vector_type(4)))  unsigned int v4u;
typedef __attribute__((ext_vector_type(8)))  int      v8i;
typedef __attribute__((ext_vector_type(4)))  int      v4i;

#define B_  256
#define TV_ 64
#define TP_ 64
#define TS_ 24
#define E_  512
#define HE_ 1024
#define H_  1024
#define A_  512

#define BM 64
#define BN 128
#define KB 32
#define LDK (KB + 8)   // 40 f16 = 80B row stride: breaks bank conflicts, keeps 16B align

#if defined(__has_builtin)
#  if __has_builtin(__builtin_amdgcn_tensor_load_to_lds) && __has_builtin(__builtin_amdgcn_s_wait_tensorcnt)
#    define USE_TDM 1
#  endif
#endif
#ifndef USE_TDM
#  define USE_TDM 0
#endif

__device__ __forceinline__ float sigmoidf_(float x) { return 1.0f / (1.0f + __expf(-x)); }

__device__ __forceinline__ unsigned int lds_off_(const void* p) {
  // LDS generic address: low 32 bits are the LDS byte offset (flat aperture rule)
  return (unsigned int)(uintptr_t)p;
}

#if USE_TDM
// ---------------------------------------------------------------------------
// Issue one TDM 2D tile load: rows_tile x KB f16 elements from a row-major
// [rows_total, K] f16 tensor starting at gptr (tile start), into LDS at
// lds_off with 4-DWORD padding every 16 DWORDs (row stride LDK f16).
// D# per CDNA5 ISA 8.3/8.4 (group0 128b, group1 256b; groups 2/3 zero = 2D).
// ---------------------------------------------------------------------------
__device__ __forceinline__ void tdm_load_tile(unsigned int lds_off, const _Float16* gptr,
                                              int K, int rows_total, int rows_tile)
{
  unsigned long long ga = (unsigned long long)(uintptr_t)gptr;
  v4u g0;
  g0[0] = 1u;                                              // count=1 (valid user D#)
  g0[1] = lds_off;                                         // lds_addr (bytes)
  g0[2] = (unsigned int)ga;                                // global_addr[31:0]
  g0[3] = (unsigned int)((ga >> 32) & 0x01FFFFFFu)         // global_addr[56:32]
        | (2u << 30);                                      // type = 2 ("image")
  unsigned int d0 = (1u << 16)                             // data_size = 2 bytes
                  | (1u << 20)                             // pad_enable
                  | (3u << 22)                             // pad_interval: 16 DWORDs
                  | (3u << 25);                            // pad_amount:   4 DWORDs
  v8i g1;
  g1[0] = (int)d0;                                         // wg_mask=0, flags
  g1[1] = (int)(((unsigned int)K & 0xFFFFu) << 16);        // tensor_dim0 [79:48] lo
  g1[2] = (int)(((unsigned int)K >> 16)
        | (((unsigned int)rows_total & 0xFFFFu) << 16));   // tensor_dim0 hi | tensor_dim1 lo
  g1[3] = (int)(((unsigned int)rows_total >> 16)
        | ((unsigned int)KB << 16));                       // tensor_dim1 hi | tile_dim0 = 32
  g1[4] = (int)(unsigned int)rows_tile;                    // tile_dim1 (tile_dim2 = 0)
  g1[5] = (int)(unsigned int)K;                            // tensor_dim0_stride lo
  g1[6] = 0;                                               // stride hi | dim1_stride lo
  g1[7] = 0;
  v4i z4 = {0, 0, 0, 0};
  v8i z8 = {0, 0, 0, 0, 0, 0, 0, 0};
  // 6-arg form on this toolchain: (g0, g1, g2, g3, g2g3_ext, cpol)
  __builtin_amdgcn_tensor_load_to_lds(g0, g1, z4, z4, z8, 0);
}
#endif

// ---------------------------------------------------------------------------
// C[M,N] = act( A[M,K](f16) * W[N,K](f16)^T  (+C if accflag) (+bias) ), C f32.
// Block tile 64x128, 256 threads = 8 waves; wave (wm 0..3, wn 0..1) owns a
// 16x64 slice -> 4 v_wmma_f32_16x16x32_f16 per 32-wide K step.
// K tiles arrive via the Tensor Data Mover, double-buffered in LDS so the DMA
// of tile k+1 overlaps the WMMA of tile k.   M%64==0, N%128==0, K%32==0.
// ---------------------------------------------------------------------------
__global__ __launch_bounds__(256)
void gemm_wmma_kernel(const _Float16* __restrict__ A, const _Float16* __restrict__ W,
                      const float* __restrict__ bias, float* __restrict__ C,
                      int M, int N, int K, int act, int accflag)
{
  __shared__ __align__(16) _Float16 sA[2][BM][LDK];
  __shared__ __align__(16) _Float16 sB[2][BN][LDK];
  const int tid  = threadIdx.x;
  const int lane = tid & 31;
  const int wave = tid >> 5;
  const int wm   = wave >> 1;                    // 0..3 : row sub-tile * 16
  const int wn   = wave & 1;                     // 0..1 : col group * 64
  const long bm  = (long)blockIdx.y * BM;
  const long bn  = (long)blockIdx.x * BN;

  v8f acc[4] = {{}, {}, {}, {}};

#if USE_TDM
  if (wave == 0) {
    tdm_load_tile(lds_off_(&sA[0][0][0]), A + bm * K, K, M, BM);
    tdm_load_tile(lds_off_(&sB[0][0][0]), W + bn * K, K, N, BN);
  }
  int cur = 0;
  for (int k0 = 0; k0 < K; k0 += KB) {
    if (wave == 0) {
      __builtin_amdgcn_s_wait_tensorcnt((short)0);       // current tiles landed
      if (k0 + KB < K) {                                  // prefetch next K tile
        tdm_load_tile(lds_off_(&sA[cur ^ 1][0][0]), A + bm * K + k0 + KB, K, M, BM);
        tdm_load_tile(lds_off_(&sB[cur ^ 1][0][0]), W + bn * K + k0 + KB, K, N, BN);
      }
    }
    __syncthreads();
#else
  int cur = 0;
  for (int k0 = 0; k0 < K; k0 += KB) {
    for (int i = tid; i < BM * KB; i += 256) {
      int r = i >> 5, c = i & (KB - 1);
      sA[0][r][c] = A[(bm + r) * (long)K + k0 + c];
    }
    for (int i = tid; i < BN * KB; i += 256) {
      int r = i >> 5, c = i & (KB - 1);
      sB[0][r][c] = W[(bn + r) * (long)K + k0 + c];
    }
    __syncthreads();
#endif
    const int half = (lane >> 4) & 1;
    // A fragment: lanes<16 hold K{0-7,16-23}, lanes>=16 hold K{8-15,24-31}
    const int ar = wm * 16 + (lane & 15);
    v8h alo = *(const v8h*)&sA[cur][ar][half * 8];
    v8h ahi = *(const v8h*)&sA[cur][ar][half * 8 + 16];
    v16h af;
    #pragma unroll
    for (int e = 0; e < 8; ++e) { af[e] = alo[e]; af[e + 8] = ahi[e]; }

    #pragma unroll
    for (int s = 0; s < 4; ++s) {                // 4 n-subtiles of 16 per wave
      const int br = wn * 64 + s * 16 + (lane & 15);
      const int kb = half * 8;
      v8h bl = *(const v8h*)&sB[cur][br][kb];
      v8h bh = *(const v8h*)&sB[cur][br][kb + 16];
      v16h bf;
      #pragma unroll
      for (int e = 0; e < 8; ++e) { bf[e] = bl[e]; bf[e + 8] = bh[e]; }
      acc[s] = __builtin_amdgcn_wmma_f32_16x16x32_f16(false, af, false, bf,
                                                      (short)0, acc[s], false, false);
    }
    __syncthreads();
#if USE_TDM
    cur ^= 1;
#endif
  }

  // epilogue per ISA C layout: lane l -> n = l&15, VGPR r -> m = r + 8*(l>=16)
  const int nl = lane & 15;
  const int mo = ((lane >> 4) & 1) * 8;
  #pragma unroll
  for (int s = 0; s < 4; ++s) {
    long gnc = bn + wn * 64 + s * 16 + nl;
    #pragma unroll
    for (int r = 0; r < 8; ++r) {
      long gm = bm + wm * 16 + mo + r;
      float v = acc[s][r];
      if (accflag) v += C[gm * N + gnc];
      if (bias)    v += bias[gnc];
      if (act == 1) v = tanhf(v);
      C[gm * N + gnc] = v;
    }
  }
}

// ---------------------------------------------------------------------------
// Bahdanau attention, one batch row per block (256 threads, 8 waves, T==64).
// Context is written as f16 straight into the fused lstm_in buffer slice.
// ---------------------------------------------------------------------------
__global__ __launch_bounds__(256)
void attention_kernel(const float* __restrict__ q, const float* __restrict__ proj,
                      const float* __restrict__ feats, const int* __restrict__ mask,
                      const float* __restrict__ Wc, const float* __restrict__ bc,
                      _Float16* __restrict__ ctx, int T, int AD, int HE,
                      int ctx_stride, int ctx_off)
{
  __shared__ float qs[512];
  __shared__ float vs[512];
  __shared__ float es[64];
  const int b    = blockIdx.x;
  const int tid  = threadIdx.x;
  const int lane = tid & 31;
  const int wave = tid >> 5;

  for (int a = tid; a < AD; a += 256) { qs[a] = q[(long)b * AD + a]; vs[a] = Wc[a]; }
  __syncthreads();
  const float bcv = bc[0];

  for (int t = wave; t < T; t += 8) {
    const float* pr = proj + ((long)b * T + t) * AD;
    float s = 0.f;
    for (int a = lane; a < AD; a += 32) s += tanhf(pr[a] + qs[a]) * vs[a];
    #pragma unroll
    for (int o = 16; o > 0; o >>= 1) s += __shfl_xor(s, o, 32);
    if (lane == 0) es[t] = mask[(long)b * T + t] ? -INFINITY : (s + bcv);
  }
  __syncthreads();

  if (wave == 0) {                 // softmax over T=64, 2 elems/lane (wave32)
    float e0 = es[lane], e1 = es[lane + 32];
    float m = fmaxf(e0, e1);
    #pragma unroll
    for (int o = 16; o > 0; o >>= 1) m = fmaxf(m, __shfl_xor(m, o, 32));
    float x0 = __expf(e0 - m), x1 = __expf(e1 - m);
    float s = x0 + x1;
    #pragma unroll
    for (int o = 16; o > 0; o >>= 1) s += __shfl_xor(s, o, 32);
    float inv = 1.0f / s;
    es[lane] = x0 * inv; es[lane + 32] = x1 * inv;
  }
  __syncthreads();

  for (int h = tid; h < HE; h += 256) {
    const float* fb = feats + (long)b * T * HE + h;
    float a = 0.f;
    for (int t = 0; t < T; ++t) a += es[t] * fb[(long)t * HE];
    ctx[(long)b * ctx_stride + ctx_off + h] = (_Float16)a;
  }
}

// ---------------------------------------------------------------------------
// LSTM pointwise: gates g[B,4H] (i,f,g,o); writes new c (f32) and h (f16,
// optionally duplicated into the attention-query state buffer).
// ---------------------------------------------------------------------------
__global__ __launch_bounds__(256)
void lstm_pointwise_kernel(const float* __restrict__ g, const float* __restrict__ c_in,
                           float* __restrict__ c_out, _Float16* __restrict__ h_out,
                           _Float16* __restrict__ h_dup, int BH, int Hd)
{
  int idx = blockIdx.x * 256 + threadIdx.x;
  if (idx >= BH) return;
  int b = idx / Hd, j = idx - b * Hd;
  long base = (long)b * 4 * Hd;
  float gi = g[base + j];
  float gf = g[base + Hd + j];
  float gg = g[base + 2 * Hd + j];
  float go = g[base + 3 * Hd + j];
  float c2 = sigmoidf_(gf) * c_in[idx] + sigmoidf_(gi) * tanhf(gg);
  float h2 = sigmoidf_(go) * tanhf(c2);
  c_out[idx] = c2;
  _Float16 hh = (_Float16)h2;
  h_out[idx] = hh;
  if (h_dup) h_dup[idx] = hh;
}

__global__ void cvt_f32_f16_kernel(const float* __restrict__ src, _Float16* __restrict__ dst,
                                   long n) {
  long i = (long)blockIdx.x * 256 + threadIdx.x;
  if (i < n) dst[i] = (_Float16)src[i];
}

// dst[r*stride + off + c] = (f16)src[r*cols + c]
__global__ void cvt_strided_kernel(const float* __restrict__ src, _Float16* __restrict__ dst,
                                   int rows, int cols, int stride, int off) {
  int i = blockIdx.x * 256 + threadIdx.x;
  if (i >= rows * cols) return;
  int r = i / cols, c = i - r * cols;
  dst[(long)r * stride + off + c] = (_Float16)src[i];
}

__global__ void add_bias_kernel(const float* __restrict__ a, const float* __restrict__ b,
                                float* __restrict__ o, int n) {
  int i = blockIdx.x * 256 + threadIdx.x;
  if (i < n) o[i] = a[i] + b[i];
}

__global__ void fill_h16_kernel(_Float16* __restrict__ o, float v, int n) {
  int i = blockIdx.x * 256 + threadIdx.x;
  if (i < n) o[i] = (_Float16)v;
}

// ---------------------------------------------------------------------------
extern "C" void kernel_launch(void* const* d_in, const int* in_sizes, int n_in,
                              void* d_out, int out_size, void* d_ws, size_t ws_size,
                              hipStream_t stream) {
  (void)in_sizes; (void)n_in; (void)out_size; (void)ws_size;
  const float* enc_hiddens = (const float*)d_in[0];
  const int*   enc_masks   = (const int*)  d_in[1];
  const float* mapped      = (const float*)d_in[2];
  const int*   p3d_masks   = (const int*)  d_in[3];
  const float* h1_init     = (const float*)d_in[4];
  const float* c1_init     = (const float*)d_in[5];
  const float* h2_init     = (const float*)d_in[6];
  const float* c2_init     = (const float*)d_in[7];
  const float* captions    = (const float*)d_in[8];
  const float* W_lstm_att  = (const float*)d_in[9];
  const float* b_lstm_att  = (const float*)d_in[10];
  const float* W_p3d_att   = (const float*)d_in[11];
  const float* b_p3d_att   = (const float*)d_in[12];
  const float* W_lstm_dec  = (const float*)d_in[13];
  const float* b_lstm_dec  = (const float*)d_in[14];
  const float* W_p3d_dec   = (const float*)d_in[15];
  const float* b_p3d_dec   = (const float*)d_in[16];
  const float* W_corr      = (const float*)d_in[17];
  const float* b_corr      = (const float*)d_in[18];
  const float* W_lt        = (const float*)d_in[19];
  const float* b_lt        = (const float*)d_in[20];
  const float* W_ih1       = (const float*)d_in[21];
  const float* W_hh1       = (const float*)d_in[22];
  const float* b_ih1       = (const float*)d_in[23];
  const float* b_hh1       = (const float*)d_in[24];
  const float* W_ih2       = (const float*)d_in[25];
  const float* W_hh2       = (const float*)d_in[26];
  const float* b_ih2       = (const float*)d_in[27];
  const float* b_hh2       = (const float*)d_in[28];
  float* out = (float*)d_out;

  // ---- workspace carve-up -------------------------------------------------
  char* p = (char*)d_ws;
  auto allocB = [&](size_t bytes) -> char* {
    char* r = p;
    p += (bytes + 255) & ~(size_t)255;
    return r;
  };
  const int XDIM = E_ + 2 * HE_;                                   // 2560
  auto allocF = [&](size_t n) -> float*     { return (float*)allocB(n * 4); };
  auto allocH = [&](size_t n) -> _Float16*  { return (_Float16*)allocB(n * 2); };

  // f16 weight copies (streamed once; whole f16 set ~52MB -> L2-resident)
  _Float16* Wh_latt = allocH((size_t)A_ * HE_);
  _Float16* Wh_patt = allocH((size_t)A_ * HE_);
  _Float16* Wh_ldec = allocH((size_t)A_ * H_);
  _Float16* Wh_pdec = allocH((size_t)A_ * H_);
  _Float16* Wh_lt   = allocH((size_t)H_ * H_);
  _Float16* Wh_ih1  = allocH((size_t)4 * H_ * XDIM);
  _Float16* Wh_hh1  = allocH((size_t)4 * H_ * H_);
  _Float16* Wh_ih2  = allocH((size_t)4 * H_ * H_);
  _Float16* Wh_hh2  = allocH((size_t)4 * H_ * H_);
  // f16 activation copies for WMMA A-operands
  _Float16* enc_h   = allocH((size_t)B_ * TV_ * HE_);
  _Float16* map_h   = allocH((size_t)B_ * TP_ * HE_);
  _Float16* lstm_in = allocH((size_t)B_ * XDIM);                   // [elem|a_t|p3d]
  _Float16* h1_h    = allocH((size_t)B_ * H_);
  _Float16* h2_h    = allocH((size_t)B_ * H_);
  _Float16* prev_h  = allocH((size_t)B_ * H_);                     // attention query
  // f32 buffers
  float* enc_proj = allocF((size_t)B_ * TV_ * A_);
  float* p3d_proj = allocF((size_t)B_ * TP_ * A_);
  float* bias1    = allocF((size_t)4 * H_);
  float* bias2    = allocF((size_t)4 * H_);
  float* q1       = allocF((size_t)B_ * A_);
  float* q2       = allocF((size_t)B_ * A_);
  float* g1       = allocF((size_t)B_ * 4 * H_);
  float* g2       = allocF((size_t)B_ * 4 * H_);
  float* c1       = allocF((size_t)B_ * H_);
  float* c2       = allocF((size_t)B_ * H_);

  auto cvt = [&](const float* s, _Float16* d, size_t n) {
    cvt_f32_f16_kernel<<<(unsigned)((n + 255) / 256), 256, 0, stream>>>(s, d, (long)n);
  };
  auto gemm = [&](const _Float16* A, const _Float16* W, const float* bias, float* C,
                  int M, int N, int K, int act, int accf) {
    dim3 grid(N / BN, M / BM);
    gemm_wmma_kernel<<<grid, 256, 0, stream>>>(A, W, bias, C, M, N, K, act, accf);
  };

  // ---- one-time pre-pass --------------------------------------------------
  cvt(W_lstm_att, Wh_latt, (size_t)A_ * HE_);
  cvt(W_p3d_att,  Wh_patt, (size_t)A_ * HE_);
  cvt(W_lstm_dec, Wh_ldec, (size_t)A_ * H_);
  cvt(W_p3d_dec,  Wh_pdec, (size_t)A_ * H_);
  cvt(W_lt,       Wh_lt,   (size_t)H_ * H_);
  cvt(W_ih1,      Wh_ih1,  (size_t)4 * H_ * XDIM);
  cvt(W_hh1,      Wh_hh1,  (size_t)4 * H_ * H_);
  cvt(W_ih2,      Wh_ih2,  (size_t)4 * H_ * H_);
  cvt(W_hh2,      Wh_hh2,  (size_t)4 * H_ * H_);
  cvt(enc_hiddens, enc_h,  (size_t)B_ * TV_ * HE_);
  cvt(mapped,      map_h,  (size_t)B_ * TP_ * HE_);
  cvt(h1_init,     h1_h,   (size_t)B_ * H_);
  cvt(h2_init,     h2_h,   (size_t)B_ * H_);

  const size_t stBytes = (size_t)B_ * H_ * 4;
  (void)hipMemcpyAsync(c1, c1_init, stBytes, hipMemcpyDeviceToDevice, stream);
  (void)hipMemcpyAsync(c2, c2_init, stBytes, hipMemcpyDeviceToDevice, stream);
  fill_h16_kernel<<<(B_ * H_ + 255) / 256, 256, 0, stream>>>(prev_h, 1.0f / (float)HE_, B_ * H_);
  add_bias_kernel<<<(4 * H_ + 255) / 256, 256, 0, stream>>>(b_ih1, b_hh1, bias1, 4 * H_);
  add_bias_kernel<<<(4 * H_ + 255) / 256, 256, 0, stream>>>(b_ih2, b_hh2, bias2, 4 * H_);

  // projections: enc_proj = enc_hiddens @ W_lstm_att^T + b (f32 out)
  gemm(enc_h, Wh_latt, b_lstm_att, enc_proj, B_ * TV_, A_, HE_, 0, 0);
  gemm(map_h, Wh_patt, b_p3d_att,  p3d_proj, B_ * TP_, A_, HE_, 0, 0);

  // ---- recurrent steps ----------------------------------------------------
  for (int t = 0; t < TS_; ++t) {
    // caption slice -> lstm_in[:, 0:E] (f16)
    cvt_strided_kernel<<<(B_ * E_ + 255) / 256, 256, 0, stream>>>(
        captions + (size_t)t * B_ * E_, lstm_in, B_, E_, XDIM, 0);

    // attention queries from prev hidden (== h2 of previous step)
    gemm(prev_h, Wh_ldec, b_lstm_dec, q1, B_, A_, H_, 0, 0);
    gemm(prev_h, Wh_pdec, b_p3d_dec,  q2, B_, A_, H_, 0, 0);

    // contexts -> lstm_in[:, E:E+HE] and lstm_in[:, E+HE:]
    attention_kernel<<<B_, 256, 0, stream>>>(q1, enc_proj, enc_hiddens, enc_masks,
                                             W_corr, b_corr, lstm_in,
                                             TV_, A_, HE_, XDIM, E_);
    attention_kernel<<<B_, 256, 0, stream>>>(q2, p3d_proj, mapped, p3d_masks,
                                             W_corr, b_corr, lstm_in,
                                             TP_, A_, HE_, XDIM, E_ + HE_);

    // layer 1: g1 = lstm_in @ W_ih1^T + h1 @ W_hh1^T + (b_ih1 + b_hh1)
    gemm(lstm_in, Wh_ih1, bias1,   g1, B_, 4 * H_, XDIM, 0, 0);
    gemm(h1_h,    Wh_hh1, nullptr, g1, B_, 4 * H_, H_,   0, 1);
    lstm_pointwise_kernel<<<(B_ * H_ + 255) / 256, 256, 0, stream>>>(
        g1, c1, c1, h1_h, nullptr, B_ * H_, H_);

    // layer 2: uses NEW h1 and OLD h2
    gemm(h1_h, Wh_ih2, bias2,   g2, B_, 4 * H_, H_, 0, 0);
    gemm(h2_h, Wh_hh2, nullptr, g2, B_, 4 * H_, H_, 0, 1);
    lstm_pointwise_kernel<<<(B_ * H_ + 255) / 256, 256, 0, stream>>>(
        g2, c2, c2, h2_h, prev_h, B_ * H_, H_);        // prev <- new h2

    // out_t = tanh(h1 @ W_lt^T + b_lt) -> d_out[t]
    gemm(h1_h, Wh_lt, b_lt, out + (size_t)t * B_ * H_, B_, H_, H_, 1, 0);
  }
}
